// BatchedPoincareBlock_60129542144391
// MI455X (gfx1250) — matrix operand order repaired and checked by reading, not compiled, not run on hardware
//
#include <hip/hip_runtime.h>
#include <hip/hip_bf16.h>

// ---------------------------------------------------------------------------
// Transformer block (GPT-2 style) for MI455X / gfx1250.
// GEMMs + attention on v_wmma_f32_16x16x32_bf16; tile staging via the
// Tensor Data Mover (tensor_load_to_lds + TENSORcnt) with double buffering.
// ---------------------------------------------------------------------------

#define BQ 2
#define SQ 2048
#define EQ 1024
#define HQ 16
#define HDQ 64
#define DFFQ 4096
#define E3Q (3 * EQ)
#define MQ (BQ * SQ)

#define BM 128
#define BN 128
#define BK 32
#define LPAD 8
#define LSTR (BK + LPAD)  // 40 ushorts = 80B per LDS row

typedef __attribute__((ext_vector_type(16))) __bf16 v16bf;
typedef __attribute__((ext_vector_type(8))) float v8f;
typedef unsigned int v4u __attribute__((ext_vector_type(4)));
typedef int v4i __attribute__((ext_vector_type(4)));
typedef int v8i __attribute__((ext_vector_type(8)));

#if defined(__has_builtin)
#if __has_builtin(__builtin_amdgcn_tensor_load_to_lds)
#define HAVE_TDM 1
#endif
#endif
#ifndef HAVE_TDM
#define HAVE_TDM 0
#endif

union FragU {
  uint4 q[2];
  v16bf v;
};

__device__ __forceinline__ unsigned short f2bf(float f) {
  unsigned int u = __float_as_uint(f);
  unsigned int r = (u + 0x7fffu + ((u >> 16) & 1u)) >> 16;
  return (unsigned short)r;
}

__device__ __forceinline__ float gelu_tanh(float x) {
  float x3 = x * x * x;
  return 0.5f * x * (1.0f + tanhf(0.7978845608028654f * (x + 0.044715f * x3)));
}

__device__ __forceinline__ v16bf load_frag2(const unsigned short* p, int off0,
                                            int off1) {
  FragU f;
  f.q[0] = *(const uint4*)(p + off0);
  f.q[1] = *(const uint4*)(p + off1);
  return f.v;
}

__device__ __forceinline__ v8f wmma_bf16(v16bf a, v16bf b, v8f c) {
  return __builtin_amdgcn_wmma_f32_16x16x32_bf16(false, a, false, b, (short)0,
                                                 c, false, false);
}

__device__ __forceinline__ void wait_tensorcnt_2() {
#if defined(__has_builtin) && __has_builtin(__builtin_amdgcn_s_wait_tensorcnt)
  __builtin_amdgcn_s_wait_tensorcnt(2);
#else
  asm volatile("s_wait_tensorcnt 2" ::: "memory");
#endif
}
__device__ __forceinline__ void wait_tensorcnt_0() {
#if defined(__has_builtin) && __has_builtin(__builtin_amdgcn_s_wait_tensorcnt)
  __builtin_amdgcn_s_wait_tensorcnt(0);
#else
  asm volatile("s_wait_tensorcnt 0" ::: "memory");
#endif
}

// Low 32 bits of a generic LDS pointer == byte offset within the workgroup's
// LDS allocation (flat LDS aperture: addr[31:0] is the LDS offset).
__device__ __forceinline__ unsigned ldsoff(const void* p) {
  return (unsigned)(unsigned long long)p;
}

#if HAVE_TDM
// Issue one TDM 2D tile load: tile_h rows x tile_w bf16 elems, source row
// stride = stride_elems, LDS destination padded per pad codes.
__device__ __forceinline__ void tdm_load_2d(unsigned lds_addr, const void* gptr,
                                            unsigned tile_w, unsigned tile_h,
                                            unsigned stride_elems,
                                            unsigned pad_interval_code,
                                            unsigned pad_amount_code) {
  unsigned long long ga = (unsigned long long)gptr;
  v4u g0;
  g0[0] = 1u;  // count=1 (valid user descriptor)
  g0[1] = lds_addr;
  g0[2] = (unsigned)ga;
  g0[3] = ((unsigned)(ga >> 32) & 0x01ffffffu) | 0x80000000u;  // type=2
  unsigned dim0 = stride_elems;  // tensor_dim0 (tile is in-bounds)
  unsigned dim1 = 1u << 20;      // tensor_dim1 (large)
  v8i g1;
  g1[0] = (int)((1u << 16)                    // data_size = 2 bytes
                | (1u << 20)                  // pad_enable
                | (pad_interval_code << 22)   // dwords before pad
                | (pad_amount_code << 25));   // pad dwords
  g1[1] = (int)((dim0 & 0xffffu) << 16);
  g1[2] = (int)(((dim0 >> 16) & 0xffffu) | ((dim1 & 0xffffu) << 16));
  g1[3] = (int)(((dim1 >> 16) & 0xffffu) | ((tile_w & 0xffffu) << 16));
  g1[4] = (int)(tile_h & 0xffffu);  // tile_dim1 ; tile_dim2 = 0
  g1[5] = (int)stride_elems;        // tensor_dim0_stride low 32
  g1[6] = 0;
  g1[7] = 0;
  v4i g2 = {0, 0, 0, 0};
  v4i g3 = {0, 0, 0, 0};
#if defined(__clang_major__) && __clang_major__ >= 23
  v8i g4 = {0, 0, 0, 0, 0, 0, 0, 0};
  __builtin_amdgcn_tensor_load_to_lds(g0, g1, g2, g3, g4, 0);
#else
  __builtin_amdgcn_tensor_load_to_lds(g0, g1, g2, g3, 0);
#endif
}
#endif

// ---------------------------------------------------------------------------
// fp32 -> bf16 conversion (weights)
// ---------------------------------------------------------------------------
__global__ void cvt_bf16_kernel(const float* __restrict__ in,
                                unsigned short* __restrict__ out, int n) {
  int i = blockIdx.x * blockDim.x + threadIdx.x;
  if (i < n) out[i] = f2bf(in[i]);
}

// ---------------------------------------------------------------------------
// LayerNorm over E=1024, one block (256 threads) per row, bf16 output.
// ---------------------------------------------------------------------------
__global__ __launch_bounds__(256) void layernorm_kernel(
    const float* __restrict__ x, const float* __restrict__ gam,
    const float* __restrict__ bet, unsigned short* __restrict__ out) {
  __shared__ __align__(16) float red[256];
  int r = blockIdx.x;
  int tid = threadIdx.x;
  const float* xr = x + (size_t)r * EQ;
  float v[4];
  float s = 0.f;
#pragma unroll
  for (int i = 0; i < 4; ++i) {
    v[i] = xr[tid + i * 256];
    s += v[i];
  }
  red[tid] = s;
  __syncthreads();
  for (int o = 128; o > 0; o >>= 1) {
    if (tid < o) red[tid] += red[tid + o];
    __syncthreads();
  }
  float mu = red[0] * (1.0f / EQ);
  __syncthreads();
  float vs = 0.f;
#pragma unroll
  for (int i = 0; i < 4; ++i) {
    float d = v[i] - mu;
    vs += d * d;
  }
  red[tid] = vs;
  __syncthreads();
  for (int o = 128; o > 0; o >>= 1) {
    if (tid < o) red[tid] += red[tid + o];
    __syncthreads();
  }
  float rstd = rsqrtf(red[0] * (1.0f / EQ) + 1e-5f);
#pragma unroll
  for (int i = 0; i < 4; ++i) {
    int c = tid + i * 256;
    out[(size_t)r * EQ + c] = f2bf((v[i] - mu) * rstd * gam[c] + bet[c]);
  }
}

// ---------------------------------------------------------------------------
// Generic bf16 GEMM: C[M,N] = A[M,K] @ W[N,K]^T + bias
// MODE 0: store bf16 ; MODE 1: gelu -> bf16 ; MODE 2: +residual -> fp32
// Block: 256 threads (8 waves). Macro tile 128x128x32, double-buffered LDS,
// tiles staged by the Tensor Data Mover when available.
// ---------------------------------------------------------------------------
template <int MODE>
__global__ __launch_bounds__(256) void gemm_bf16_kernel(
    const unsigned short* __restrict__ A, const unsigned short* __restrict__ W,
    const float* __restrict__ bias, int M, int N, int K,
    const float* __restrict__ resid, float* __restrict__ outF,
    unsigned short* __restrict__ outB) {
  __shared__ __align__(16) unsigned short As[2][BM][LSTR];
  __shared__ __align__(16) unsigned short Ws[2][BN][LSTR];

  int tid = threadIdx.x;
  int lane = tid & 31;
  int wave = tid >> 5;
  int n0 = blockIdx.x * BN;
  int m0 = blockIdx.y * BM;
  int wm = (wave >> 1) * 32;  // 0,32,64,96
  int wn = (wave & 1) * 64;   // 0,64

  v8f acc[2][4] = {};

  int row = lane & 15;
  int hi = lane >> 4;
  int a0 = hi * 16;  // A-fragment K base (contiguous, matches B)
  int b0 = hi * 16;  // B-fragment K base (contiguous, matches A)

  const int NT = K / BK;

  auto issue = [&](int t) {
    int buf = t & 1;
    int k0 = t * BK;
#if HAVE_TDM
    if (wave == 0) {
      tdm_load_2d(ldsoff(&As[buf][0][0]), A + (size_t)m0 * K + k0, BK, BM, K,
                  /*interval 16dw*/ 3, /*amount 4dw*/ 3);
      tdm_load_2d(ldsoff(&Ws[buf][0][0]), W + (size_t)n0 * K + k0, BK, BN, K,
                  3, 3);
    }
#else
    int r = tid >> 1, c = (tid & 1) * 16;
    const uint4* sa = (const uint4*)(A + (size_t)(m0 + r) * K + k0 + c);
    *(uint4*)&As[buf][r][c] = sa[0];
    *(uint4*)&As[buf][r][c + 8] = sa[1];
    const uint4* sw = (const uint4*)(W + (size_t)(n0 + r) * K + k0 + c);
    *(uint4*)&Ws[buf][r][c] = sw[0];
    *(uint4*)&Ws[buf][r][c + 8] = sw[1];
#endif
  };

  issue(0);
  issue(1);

  for (int t = 0; t < NT; ++t) {
#if HAVE_TDM
    if (wave == 0) {
      if (t == NT - 1)
        wait_tensorcnt_0();  // last tile: drain everything
      else
        wait_tensorcnt_2();  // tile t done, tile t+1 may remain in flight
    }
#endif
    __syncthreads();
    int buf = t & 1;
    v16bf af[2], bf[4];
#pragma unroll
    for (int i = 0; i < 2; ++i)
      af[i] = load_frag2(&As[buf][wm + i * 16 + row][0], a0, a0 + 8);
#pragma unroll
    for (int j = 0; j < 4; ++j)
      bf[j] = load_frag2(&Ws[buf][wn + j * 16 + row][0], b0, b0 + 8);
#pragma unroll
    for (int i = 0; i < 2; ++i)
#pragma unroll
      for (int j = 0; j < 4; ++j)
        acc[i][j] = wmma_bf16(af[i], bf[j], acc[i][j]);
    __syncthreads();
    if (t + 2 < NT) issue(t + 2);
  }

  // Epilogue: C layout -> lane<16: n=lane, m=g ; lane>=16: n=lane-16, m=8+g
#pragma unroll
  for (int i = 0; i < 2; ++i)
#pragma unroll
    for (int j = 0; j < 4; ++j) {
      int n = n0 + wn + j * 16 + row;
      float bn = bias[n];
#pragma unroll
      for (int g = 0; g < 8; ++g) {
        int m = m0 + wm + i * 16 + hi * 8 + g;
        float v = acc[i][j][g] + bn;
        size_t idx = (size_t)m * N + n;
        if (MODE == 0) {
          outB[idx] = f2bf(v);
        } else if (MODE == 1) {
          outB[idx] = f2bf(gelu_tanh(v));
        } else {
          outF[idx] = v + resid[idx];
        }
      }
    }
}

// ---------------------------------------------------------------------------
// Causal flash attention. qkv: bf16 [B*S][3E], per head h columns
// [h*192 .. +63]=Q, +64=K, +128=V. Output attn bf16 [B*S][E].
// Grid: B*H*(S/64) blocks of 128 threads; wave w owns 16 queries.
// K tile staged via TDM when available; V transposed manually into LDS.
// ---------------------------------------------------------------------------
__global__ __launch_bounds__(128) void attention_kernel(
    const unsigned short* __restrict__ qkv, unsigned short* __restrict__ attn) {
  __shared__ __align__(16) unsigned short Klds[32][64 + LPAD];
  __shared__ __align__(16) unsigned short Vt[64][32 + LPAD];
  __shared__ __align__(16) unsigned short Plds[4][16][32 + LPAD];

  int tid = threadIdx.x;
  int lane = tid & 31;
  int wave = tid >> 5;
  int blk = blockIdx.x;
  int qt = blk % (SQ / 64);
  int bh = blk / (SQ / 64);
  int h = bh % HQ;
  int b = bh / HQ;
  int q0 = qt * 64 + wave * 16;

  int row = lane & 15;
  int hi = lane >> 4;
  int a0 = hi * 16;
  int b0 = hi * 16;

  const unsigned short* qbase =
      qkv + (size_t)(b * SQ + q0 + row) * E3Q + h * (3 * HDQ);
  v16bf qf0 = load_frag2(qbase, a0, a0 + 8);       // d = 0..31
  v16bf qf1 = load_frag2(qbase + 32, a0, a0 + 8);  // d = 32..63

  v8f o[4] = {};
  float mrun[8], lrun[8];
#pragma unroll
  for (int g = 0; g < 8; ++g) {
    mrun[g] = -1e30f;
    lrun[g] = 0.f;
  }

  int kend = qt * 64 + 64;  // causal upper bound for this workgroup
  for (int kc = 0; kc < kend; kc += 32) {
    __syncthreads();
#if HAVE_TDM
    if (wave == 0) {
      tdm_load_2d(ldsoff(&Klds[0][0]),
                  qkv + (size_t)(b * SQ + kc) * E3Q + h * (3 * HDQ) + HDQ,
                  /*tile_w*/ 64, /*tile_h*/ 32, /*stride*/ E3Q,
                  /*interval 32dw*/ 4, /*amount 4dw*/ 3);
    }
#endif
    {
      int r = tid >> 2, c = (tid & 3) * 16;
#if !HAVE_TDM
      const unsigned short* ks =
          qkv + (size_t)(b * SQ + kc + r) * E3Q + h * (3 * HDQ) + HDQ + c;
      *(uint4*)&Klds[r][c] = *(const uint4*)ks;
      *(uint4*)&Klds[r][c + 8] = *(const uint4*)(ks + 8);
#endif
      const unsigned short* vs =
          qkv + (size_t)(b * SQ + kc + r) * E3Q + h * (3 * HDQ) + 2 * HDQ + c;
      uint4 p0 = *(const uint4*)vs;
      uint4 p1 = *(const uint4*)(vs + 8);
      unsigned short tmp[16];
      *(uint4*)&tmp[0] = p0;
      *(uint4*)&tmp[8] = p1;
#pragma unroll
      for (int j = 0; j < 16; ++j) Vt[c + j][r] = tmp[j];  // transpose V
    }
#if HAVE_TDM
    if (wave == 0) wait_tensorcnt_0();
#endif
    __syncthreads();

    // scores for two 16-key subtiles
    float s[2][8];
#pragma unroll
    for (int j = 0; j < 2; ++j) {
      v16bf kf0 = load_frag2(&Klds[j * 16 + row][0], b0, b0 + 8);
      v16bf kf1 = load_frag2(&Klds[j * 16 + row][32], b0, b0 + 8);
      v8f sc = {};
      sc = wmma_bf16(qf0, kf0, sc);
      sc = wmma_bf16(qf1, kf1, sc);
      int key = kc + j * 16 + row;
#pragma unroll
      for (int g = 0; g < 8; ++g) {
        int qrow = q0 + hi * 8 + g;
        float v = sc[g] * 0.125f;  // 1/sqrt(64)
        s[j][g] = (key <= qrow) ? v : -1e30f;
      }
    }

    // online softmax (row reductions across 16-lane half-groups)
    float mnew[8], alpha[8];
#pragma unroll
    for (int g = 0; g < 8; ++g) {
      float mx = fmaxf(s[0][g], s[1][g]);
#pragma unroll
      for (int msk = 1; msk < 16; msk <<= 1)
        mx = fmaxf(mx, __shfl_xor(mx, msk, 32));
      mnew[g] = fmaxf(mrun[g], mx);
      alpha[g] = __expf(mrun[g] - mnew[g]);
      mrun[g] = mnew[g];
    }
#pragma unroll
    for (int g = 0; g < 8; ++g) {
      float p0 = __expf(s[0][g] - mnew[g]);
      float p1 = __expf(s[1][g] - mnew[g]);
      float sm = p0 + p1;
#pragma unroll
      for (int msk = 1; msk < 16; msk <<= 1) sm += __shfl_xor(sm, msk, 32);
      lrun[g] = lrun[g] * alpha[g] + sm;
      Plds[wave][hi * 8 + g][row] = f2bf(p0);
      Plds[wave][hi * 8 + g][16 + row] = f2bf(p1);
    }
#pragma unroll
    for (int nt = 0; nt < 4; ++nt)
#pragma unroll
      for (int g = 0; g < 8; ++g) o[nt][g] *= alpha[g];

    asm volatile("s_wait_dscnt 0" ::: "memory");

    v16bf pf = load_frag2(&Plds[wave][row][0], a0, a0 + 8);
#pragma unroll
    for (int nt = 0; nt < 4; ++nt) {
      v16bf vf = load_frag2(&Vt[nt * 16 + row][0], b0, b0 + 8);
      o[nt] = wmma_bf16(pf, vf, o[nt]);
    }
  }

#pragma unroll
  for (int nt = 0; nt < 4; ++nt)
#pragma unroll
    for (int g = 0; g < 8; ++g) {
      int m = q0 + hi * 8 + g;
      int d = nt * 16 + row;
      float v = o[nt][g] / lrun[g];
      attn[(size_t)(b * SQ + m) * EQ + h * HDQ + d] = f2bf(v);
    }
}

// ---------------------------------------------------------------------------
// Host-side orchestration
// ---------------------------------------------------------------------------
static inline size_t align256(size_t x) { return (x + 255) & ~(size_t)255; }

extern "C" void kernel_launch(void* const* d_in, const int* in_sizes, int n_in,
                              void* d_out, int out_size, void* d_ws,
                              size_t ws_size, hipStream_t stream) {
  (void)in_sizes;
  (void)n_in;
  (void)out_size;
  (void)ws_size;

  const float* x = (const float*)d_in[0];
  const float* ln1_g = (const float*)d_in[1];
  const float* ln1_b = (const float*)d_in[2];
  const float* ln2_g = (const float*)d_in[3];
  const float* ln2_b = (const float*)d_in[4];
  const float* w_qkv = (const float*)d_in[5];
  const float* b_qkv = (const float*)d_in[6];
  const float* w_ao = (const float*)d_in[7];
  const float* b_ao = (const float*)d_in[8];
  const float* w_fc = (const float*)d_in[9];
  const float* b_fc = (const float*)d_in[10];
  const float* w_po = (const float*)d_in[11];
  const float* b_po = (const float*)d_in[12];
  float* out = (float*)d_out;

  char* ws = (char*)d_ws;
  size_t off = 0;
  auto takeU16 = [&](size_t elems) {
    unsigned short* p = (unsigned short*)(ws + off);
    off = align256(off + elems * sizeof(unsigned short));
    return p;
  };
  auto takeF32 = [&](size_t elems) {
    float* p = (float*)(ws + off);
    off = align256(off + elems * sizeof(float));
    return p;
  };

  unsigned short* wqkv_bf = takeU16((size_t)3 * EQ * EQ);
  unsigned short* wao_bf = takeU16((size_t)EQ * EQ);
  unsigned short* wfc_bf = takeU16((size_t)DFFQ * EQ);
  unsigned short* wpo_bf = takeU16((size_t)EQ * DFFQ);
  unsigned short* h1_bf = takeU16((size_t)MQ * EQ);
  unsigned short* qkv_bf = takeU16((size_t)MQ * E3Q);
  unsigned short* attn_bf = takeU16((size_t)MQ * EQ);
  float* x1_f = takeF32((size_t)MQ * EQ);
  unsigned short* h2_bf = takeU16((size_t)MQ * EQ);
  unsigned short* ffg_bf = takeU16((size_t)MQ * DFFQ);

  {
    int n1 = 3 * EQ * EQ, n2 = EQ * EQ, n3 = DFFQ * EQ, n4 = EQ * DFFQ;
    cvt_bf16_kernel<<<(n1 + 255) / 256, 256, 0, stream>>>(w_qkv, wqkv_bf, n1);
    cvt_bf16_kernel<<<(n2 + 255) / 256, 256, 0, stream>>>(w_ao, wao_bf, n2);
    cvt_bf16_kernel<<<(n3 + 255) / 256, 256, 0, stream>>>(w_fc, wfc_bf, n3);
    cvt_bf16_kernel<<<(n4 + 255) / 256, 256, 0, stream>>>(w_po, wpo_bf, n4);
  }

  layernorm_kernel<<<MQ, 256, 0, stream>>>(x, ln1_g, ln1_b, h1_bf);

  gemm_bf16_kernel<0><<<dim3(E3Q / BN, MQ / BM), 256, 0, stream>>>(
      h1_bf, wqkv_bf, b_qkv, MQ, E3Q, EQ, nullptr, nullptr, qkv_bf);

  attention_kernel<<<BQ * HQ * (SQ / 64), 128, 0, stream>>>(qkv_bf, attn_bf);

  gemm_bf16_kernel<2><<<dim3(EQ / BN, MQ / BM), 256, 0, stream>>>(
      attn_bf, wao_bf, b_ao, MQ, EQ, EQ, x, x1_f, nullptr);

  layernorm_kernel<<<MQ, 256, 0, stream>>>(x1_f, ln2_g, ln2_b, h2_bf);

  gemm_bf16_kernel<1><<<dim3(DFFQ / BN, MQ / BM), 256, 0, stream>>>(
      h2_bf, wfc_bf, b_fc, MQ, DFFQ, EQ, nullptr, nullptr, ffg_bf);

  gemm_bf16_kernel<2><<<dim3(EQ / BN, MQ / BM), 256, 0, stream>>>(
      ffg_bf, wpo_bf, b_po, MQ, EQ, DFFQ, x1_f, out, nullptr);
}
